// ClusterAttention_79654463472256
// MI455X (gfx1250) — matrix operand order repaired and checked by reading, not compiled, not run on hardware
//
#include <hip/hip_runtime.h>
#include <hip/hip_bf16.h>

// MI455X / gfx1250, wave32. Matrix math via v_wmma_f32_16x16x32_bf16;
// main GEMM uses async global->LDS (ASYNCcnt) double-buffered A staging.

typedef __attribute__((ext_vector_type(16))) __bf16 v16bf;
typedef __attribute__((ext_vector_type(8)))  float  v8f;

#define N_B   32
#define C_DIM 256
#define P_DIM 2048
#define K_CL  16
#define TAPS  15

// float -> bf16 round-to-nearest-even
__device__ __forceinline__ unsigned short f2bf(float f) {
  union { float f; unsigned int u; } c; c.f = f;
  unsigned int u = c.u;
  u += 0x7FFFu + ((u >> 16) & 1u);
  return (unsigned short)(u >> 16);
}

// A fragment (16x32 bf16): lane holds row m=lane&15; K in two runs of 8:
//   lanes 0-15: K 0-7 then 16-23 ; lanes 16-31: K 8-15 then 24-31  (klo = 0 or 8)
__device__ __forceinline__ v16bf ld_fragA(const unsigned short* line, int klo) {
  union { v16bf v; uint4 q[2]; } u;
  u.q[0] = *(const uint4*)(line + klo);
  u.q[1] = *(const uint4*)(line + 16 + klo);
  return u.v;
}
// B fragment (32x16 bf16): lane holds col n=lane&15; K in two runs of 8:
//   lanes 0-15: K 0-7 then 16-23 ; lanes 16-31: K 8-15 then 24-31   (kb = 0 or 16)
__device__ __forceinline__ v16bf ld_fragB(const unsigned short* line, int kb) {
  union { v16bf v; uint4 q[2]; } u;
  u.q[0] = *(const uint4*)(line + (kb >> 1));
  u.q[1] = *(const uint4*)(line + 16 + (kb >> 1));
  return u.v;
}

// ---------------------------------------------------------------------------
// K0: w_post f32 -> bf16 (so main GEMM A-staging is a pure async b128 copy)
// ---------------------------------------------------------------------------
__global__ __launch_bounds__(256)
void k_cvtw(const float* __restrict__ w_post, unsigned short* __restrict__ wp_bf) {
  const int total = C_DIM * K_CL * C_DIM;            // 256 * 4096
  for (int i = blockIdx.x * 256 + threadIdx.x; i < total; i += gridDim.x * 256)
    wp_bf[i] = f2bf(w_post[i]);
}

// ---------------------------------------------------------------------------
// K1: bm1[n,k,c] = conv_c( x^T, w2 ) + b2 = sum_t shift_c( W2_t @ X_n^T ) + b2
// One WG: one n, 64 c-columns (4 waves, one 16-col tile each).
// ---------------------------------------------------------------------------
__global__ __launch_bounds__(128)
void k_bm1(const float* __restrict__ x, const float* __restrict__ w2,
           const float* __restrict__ b2, float* __restrict__ bm1) {
  __shared__ unsigned short xs[78 * 264];   // [c' row][p' col] bf16
  __shared__ unsigned short w2s[16 * 264];  // [k][p' col] bf16 for current tap
  const int tid = threadIdx.x;
  const int lane = tid & 31;
  const int wv = tid >> 5;
  const int c0 = blockIdx.x * 64;
  const int n  = blockIdx.y;
  const int ln = lane & 15;
  const int klo = (lane >= 16) ? 8 : 0;
  const int kbB = (lane >= 16) ? 16 : 0;

  v8f acc = {};

  for (int pb = 0; pb < 8; ++pb) {
    const int pbase = pb * 256;
    __syncthreads();
    for (int i = tid; i < 78 * 256; i += 128) {
      int row = i >> 8, j = i & 255;
      int cp = c0 - 7 + row;
      float v = 0.f;
      if (cp >= 0 && cp < C_DIM)
        v = x[((size_t)n * C_DIM + cp) * P_DIM + pbase + j];
      xs[row * 264 + j] = f2bf(v);
    }
    for (int t = 0; t < TAPS; ++t) {
      __syncthreads();
      for (int i = tid; i < 16 * 256; i += 128) {
        int kk = i >> 8, j = i & 255;
        w2s[kk * 264 + j] = f2bf(w2[((size_t)kk * P_DIM + pbase + j) * TAPS + t]);
      }
      __syncthreads();
      const unsigned short* aline = w2s + ln * 264;
      const unsigned short* bline = xs + (16 * wv + ln + t) * 264;
      for (int kc = 0; kc < 8; ++kc) {
        v16bf a = ld_fragA(aline + kc * 32, klo);
        v16bf b = ld_fragB(bline + kc * 32, kbB);
        acc = __builtin_amdgcn_wmma_f32_16x16x32_bf16(false, a, false, b,
                                                      (short)0, acc, false, false);
      }
    }
  }
  const int col = c0 + 16 * wv + ln;
  const int mhi = (lane >= 16) ? 8 : 0;
  for (int r = 0; r < 8; ++r) {
    int m = r + mhi;
    bm1[((size_t)n * K_CL + m) * C_DIM + col] = acc[r] + b2[m];
  }
}

// ---------------------------------------------------------------------------
// K2: scores = conv_p(x, w1) + b1 + bm1 @ x ; probs = softmax_k(scores)
// One WG: one n, 64 p-columns (4 waves). x tile staged once as [p][c].
// ---------------------------------------------------------------------------
__global__ __launch_bounds__(128)
void k_probs(const float* __restrict__ x, const float* __restrict__ w1,
             const float* __restrict__ b1, const float* __restrict__ bm1,
             float* __restrict__ probs) {
  __shared__ unsigned short xs[78 * 264];   // [p row][c col] bf16
  __shared__ unsigned short w1s[16 * 264];
  __shared__ float sm[16 * 64];
  const int tid = threadIdx.x;
  const int lane = tid & 31;
  const int wv = tid >> 5;
  const int p0 = blockIdx.x * 64;
  const int n  = blockIdx.y;
  const int ln = lane & 15;
  const int klo = (lane >= 16) ? 8 : 0;
  const int kbB = (lane >= 16) ? 16 : 0;

  v8f acc = {};

  for (int i = tid; i < 256 * 80; i += 128) {
    int cc = i / 80, pp = i % 80;
    if (pp < 78) {
      int p = p0 - 7 + pp;
      float v = 0.f;
      if (p >= 0 && p < P_DIM)
        v = x[((size_t)n * C_DIM + cc) * P_DIM + p];
      xs[pp * 264 + cc] = f2bf(v);
    }
  }

  for (int t = 0; t < TAPS; ++t) {
    __syncthreads();
    for (int i = tid; i < 16 * 256; i += 128) {
      int kk = i >> 8, cc = i & 255;
      w1s[kk * 264 + cc] = f2bf(w1[((size_t)kk * C_DIM + cc) * TAPS + t]);
    }
    __syncthreads();
    const unsigned short* aline = w1s + ln * 264;
    const unsigned short* bline = xs + (16 * wv + ln + t) * 264;
    for (int kc = 0; kc < 8; ++kc) {
      v16bf a = ld_fragA(aline + kc * 32, klo);
      v16bf b = ld_fragB(bline + kc * 32, kbB);
      acc = __builtin_amdgcn_wmma_f32_16x16x32_bf16(false, a, false, b,
                                                    (short)0, acc, false, false);
    }
  }

  __syncthreads();
  for (int i = tid; i < 16 * 256; i += 128) {
    int kk = i >> 8, cc = i & 255;
    w1s[kk * 264 + cc] = f2bf(bm1[((size_t)n * K_CL + kk) * C_DIM + cc]);
  }
  __syncthreads();
  {
    const unsigned short* aline = w1s + ln * 264;
    const unsigned short* bline = xs + (16 * wv + ln + 7) * 264;
    for (int kc = 0; kc < 8; ++kc) {
      v16bf a = ld_fragA(aline + kc * 32, klo);
      v16bf b = ld_fragB(bline + kc * 32, kbB);
      acc = __builtin_amdgcn_wmma_f32_16x16x32_bf16(false, a, false, b,
                                                    (short)0, acc, false, false);
    }
  }

  const int mhi = (lane >= 16) ? 8 : 0;
  for (int r = 0; r < 8; ++r)
    sm[(r + mhi) * 64 + 16 * wv + ln] = acc[r];
  __syncthreads();
  if (tid < 64) {
    float vbuf[16];
    float mx = -1e30f;
    for (int kk = 0; kk < 16; ++kk) {
      float v = sm[kk * 64 + tid] + b1[kk];
      vbuf[kk] = v;
      mx = fmaxf(mx, v);
    }
    float s = 0.f;
    for (int kk = 0; kk < 16; ++kk) { vbuf[kk] = __expf(vbuf[kk] - mx); s += vbuf[kk]; }
    float inv = 1.0f / s;
    for (int kk = 0; kk < 16; ++kk)
      probs[((size_t)n * K_CL + kk) * P_DIM + p0 + tid] = vbuf[kk] * inv;
  }
}

// ---------------------------------------------------------------------------
// K3: out[n,o,p] = sum_k probs[n,k,p] * (W_k @ X)[o,p] + b_post[o]
// Probs applied at ACCUMULATOR level (per C-fragment column) so the x tile is
// staged once per WG; per-chunk w_post staging is async b128 global->LDS,
// double-buffered, overlapped with the WMMA chain.
// WG = 256 thr (8 waves): 128o x 64p tile; wave = 32o x 32p (2x2 fragments).
// ---------------------------------------------------------------------------
__global__ __launch_bounds__(256)
void k_out(const float* __restrict__ x, const unsigned short* __restrict__ wp_bf,
           const float* __restrict__ b_post, const float* __restrict__ probs,
           float* __restrict__ out) {
  __shared__ unsigned short Bs[64 * 264];    // [p row][c 0..255] bf16, staged once
  __shared__ unsigned short As[2][128 * 40]; // double-buffered w_post chunk
  const int tid = threadIdx.x;
  const int lane = tid & 31;
  const int wv = tid >> 5;
  const int wo = wv & 3;                     // 32-row o group
  const int wp = wv >> 2;                    // 32-col p group
  const int p0 = blockIdx.x * 64;
  const int o0 = blockIdx.y * 128;
  const int n  = blockIdx.z;
  const int ln = lane & 15;
  const int klo = (lane >= 16) ? 8 : 0;
  const int kbB = (lane >= 16) ? 16 : 0;

  // stage x tile once: Bs[pp][cc] = bf16(x[n][cc][p0+pp])
  for (int i = tid; i < 64 * 256; i += 256) {
    int cc = i >> 6, pp = i & 63;
    Bs[pp * 264 + cc] = f2bf(x[((size_t)n * C_DIM + cc) * P_DIM + p0 + pp]);
  }

  // async global->LDS copy of one 128x32 bf16 w_post chunk (512 b128 ops)
  auto stageA = [&](int buf, int kcl, int cb) {
    unsigned base = (unsigned)(size_t)(&As[buf][0]);
    for (int s = 0; s < 2; ++s) {
      int idx = tid + 256 * s;
      int row = idx >> 2, seg = idx & 3;
      unsigned loff = base + row * 80 + seg * 16;
      const unsigned short* g =
          wp_bf + ((size_t)(o0 + row) << 12) + (kcl << 8) + cb + (seg << 3);
      asm volatile("global_load_async_to_lds_b128 %0, %1, off"
                   :: "v"(loff), "v"(g) : "memory");
    }
  };

  stageA(0, 0, 0);  // prologue

  v8f macc[2][2] = {};
  for (int kcl = 0; kcl < K_CL; ++kcl) {
    const float pk0 = probs[((size_t)n * K_CL + kcl) * P_DIM + p0 + 32 * wp + ln];
    const float pk1 = probs[((size_t)n * K_CL + kcl) * P_DIM + p0 + 32 * wp + 16 + ln];
    v8f tacc[2][2] = {};
    for (int cbi = 0; cbi < 8; ++cbi) {
      const int kc = kcl * 8 + cbi;
      asm volatile("s_wait_asynccnt 0" ::: "memory");
      __syncthreads();
      if (kc + 1 < 128)
        stageA((kc + 1) & 1, (kc + 1) >> 3, ((kc + 1) & 7) << 5);
      const unsigned short* Ab = As[kc & 1];
      const int cb = cbi << 5;
      v16bf afr[2], bfr[2];
      for (int mt = 0; mt < 2; ++mt)
        afr[mt] = ld_fragA(Ab + (32 * wo + 16 * mt + ln) * 40, klo);
      for (int nt = 0; nt < 2; ++nt)
        bfr[nt] = ld_fragB(Bs + (32 * wp + 16 * nt + ln) * 264 + cb, kbB);
      for (int mt = 0; mt < 2; ++mt)
        for (int nt = 0; nt < 2; ++nt)
          tacc[mt][nt] = __builtin_amdgcn_wmma_f32_16x16x32_bf16(
              false, afr[mt], false, bfr[nt], (short)0, tacc[mt][nt], false, false);
    }
    for (int mt = 0; mt < 2; ++mt)
      for (int nt = 0; nt < 2; ++nt) {
        const float pk = nt ? pk1 : pk0;
        for (int r = 0; r < 8; ++r)
          macc[mt][nt][r] += tacc[mt][nt][r] * pk;
      }
  }

  const int mhi = (lane >= 16) ? 8 : 0;
  for (int mt = 0; mt < 2; ++mt)
    for (int nt = 0; nt < 2; ++nt)
      for (int r = 0; r < 8; ++r) {
        int m = o0 + 32 * wo + 16 * mt + r + mhi;
        int col = p0 + 32 * wp + 16 * nt + ln;
        out[((size_t)n * C_DIM + m) * P_DIM + col] = macc[mt][nt][r] + b_post[m];
      }
}

// ---------------------------------------------------------------------------
extern "C" void kernel_launch(void* const* d_in, const int* in_sizes, int n_in,
                              void* d_out, int out_size, void* d_ws, size_t ws_size,
                              hipStream_t stream) {
  const float* x      = (const float*)d_in[0];
  const float* w1     = (const float*)d_in[1];
  const float* b1     = (const float*)d_in[2];
  const float* w2     = (const float*)d_in[3];
  const float* b2     = (const float*)d_in[4];
  const float* w_post = (const float*)d_in[5];
  const float* b_post = (const float*)d_in[6];
  float* out = (float*)d_out;

  float* probs = (float*)d_ws;                                // 4 MB
  float* bm1   = probs + (size_t)N_B * K_CL * P_DIM;          // 512 KB
  unsigned short* wpbf =
      (unsigned short*)(bm1 + (size_t)N_B * K_CL * C_DIM);    // 2 MB

  k_cvtw <<<dim3(1024),       256, 0, stream>>>(w_post, wpbf);
  k_bm1  <<<dim3(4, N_B),     128, 0, stream>>>(x, w2, b2, bm1);
  k_probs<<<dim3(32, N_B),    128, 0, stream>>>(x, w1, b1, bm1, probs);
  k_out  <<<dim3(32, 2, N_B), 256, 0, stream>>>(x, wpbf, b_post, probs, out);
}
